// CrossModel_31061203485126
// MI455X (gfx1250) — compile-verified, matrix operand order checked by reading, not run on hardware
//
#include <hip/hip_runtime.h>
#include <math.h>

#define N_ENTITY  64368
#define N_CONCEPT 29309
#define DIM       128
#define NUM_BASES 8
#define RD_DB     12
#define RD_RG     8
#define NE_DB     400000
#define NE_RG     200000
#define NE_CON    150000
#define B_SZ      128
#define L_DB      64
#define L_CON     128
#define NEG_INF_F (-1e30f)

typedef __attribute__((ext_vector_type(2)))  float   v2f;
typedef __attribute__((ext_vector_type(8)))  float   v8f;
typedef __attribute__((ext_vector_type(16))) __bf16  v16bf;

#if __has_builtin(__builtin_amdgcn_wmma_f32_16x16x4_f32)
#define HAVE_WMMA_F32X4 1
#endif

// ---------------------------------------------------------------------------
// WMMA fragment helpers.  Both A and B operands are loaded from "row
// pointers": for a 16x16 output tile, lane&15 selects the M-row (A) / N-row
// (B, stored N-major so per-lane loads are contiguous along K).
// ---------------------------------------------------------------------------
#if HAVE_WMMA_F32X4
#define KSTEP 4
typedef v2f frag_t;
__device__ __forceinline__ frag_t cm_load_frag(const float* __restrict__ row,
                                               int k, int lane) {
    int koff = (lane >> 4) << 1;        // lanes 0-15: K 0,1 ; lanes 16-31: K 2,3
    frag_t f;
    f.x = row[k + koff];
    f.y = row[k + koff + 1];
    return f;
}
__device__ __forceinline__ v8f cm_mac(frag_t a, frag_t b, v8f c) {
    return __builtin_amdgcn_wmma_f32_16x16x4_f32(false, a, false, b,
                                                 (short)0, c, false, false);
}
#else
#define KSTEP 32
typedef v16bf frag_t;
__device__ __forceinline__ frag_t cm_load_frag(const float* __restrict__ row,
                                               int k, int lane) {
    int kbase = (lane >> 4) * 8;
    frag_t f;
    #pragma unroll
    for (int j = 0; j < 16; ++j) {
        int kk = k + ((j < 8) ? 0 : 16) + kbase + (j & 7);
        f[j] = (__bf16)row[kk];
    }
    return f;
}
__device__ __forceinline__ v8f cm_mac(frag_t a, frag_t b, v8f c) {
    return __builtin_amdgcn_wmma_f32_16x16x32_bf16(false, a, false, b,
                                                   (short)0, c, false, false);
}
#endif

// ---------------------------------------------------------------------------
// RGCN path A (fused): one wave per edge, msg = sum_b comp[et,b]*basis[b,src,:]
// ---------------------------------------------------------------------------
__global__ void cm_rgcn_edge(const int* __restrict__ src, const int* __restrict__ dst,
                             const int* __restrict__ et,
                             const float* __restrict__ basis,
                             const float* __restrict__ comp,
                             float* __restrict__ agg, float* __restrict__ deg, int nE) {
    int wave = threadIdx.x >> 5;
    int lane = threadIdx.x & 31;
    int e = blockIdx.x * 8 + wave;
    if (e >= nE) return;
    int s = src[e], dv = dst[e], r = et[e];
    const size_t ND = (size_t)N_ENTITY * DIM;
    int d0 = lane * 4;
    float4 acc = make_float4(0.f, 0.f, 0.f, 0.f);
    #pragma unroll
    for (int b = 0; b < NUM_BASES; ++b) {
        float cw = comp[r * NUM_BASES + b];
        const float4 v = *(const float4*)(basis + b * ND + (size_t)s * DIM + d0);
        acc.x += cw * v.x; acc.y += cw * v.y; acc.z += cw * v.z; acc.w += cw * v.w;
    }
    float* o = agg + (size_t)dv * DIM + d0;
    atomicAdd(o + 0, acc.x); atomicAdd(o + 1, acc.y);
    atomicAdd(o + 2, acc.z); atomicAdd(o + 3, acc.w);
    if (lane == 0) atomicAdd(deg + dv, 1.0f);
}

// ---------------------------------------------------------------------------
// RGCN path B (materialized): weight[r,n,:] = sum_b comp[r,b]*basis[b,n,:]
// (basis read once, then 512B gather per edge) -- used when ws is big enough.
// ---------------------------------------------------------------------------
__global__ void cm_make_weight(const float* __restrict__ basis,
                               const float* __restrict__ comp,
                               float* __restrict__ weight, int R) {
    const size_t ND = (size_t)N_ENTITY * DIM;
    size_t q = (size_t)blockIdx.x * blockDim.x + threadIdx.x;   // float4 index
    if (q >= ND / 4) return;
    size_t o = q * 4;
    float4 bs[NUM_BASES];
    #pragma unroll
    for (int b = 0; b < NUM_BASES; ++b)
        bs[b] = *(const float4*)(basis + b * ND + o);
    for (int r = 0; r < R; ++r) {
        float4 acc = make_float4(0.f, 0.f, 0.f, 0.f);
        #pragma unroll
        for (int b = 0; b < NUM_BASES; ++b) {
            float cw = comp[r * NUM_BASES + b];
            acc.x += cw * bs[b].x; acc.y += cw * bs[b].y;
            acc.z += cw * bs[b].z; acc.w += cw * bs[b].w;
        }
        *(float4*)(weight + (size_t)r * ND + o) = acc;
    }
}

__global__ void cm_rgcn_edge_w(const int* __restrict__ src, const int* __restrict__ dst,
                               const int* __restrict__ et,
                               const float* __restrict__ weight,
                               float* __restrict__ agg, float* __restrict__ deg, int nE) {
    int wave = threadIdx.x >> 5;
    int lane = threadIdx.x & 31;
    int e = blockIdx.x * 8 + wave;
    if (e >= nE) return;
    int s = src[e], dv = dst[e], r = et[e];
    int d0 = lane * 4;
    const float4 v = *(const float4*)(weight +
        ((size_t)r * N_ENTITY + (size_t)s) * DIM + d0);
    float* o = agg + (size_t)dv * DIM + d0;
    atomicAdd(o + 0, v.x); atomicAdd(o + 1, v.y);
    atomicAdd(o + 2, v.z); atomicAdd(o + 3, v.w);
    if (lane == 0) atomicAdd(deg + dv, 1.0f);
}

__global__ void cm_entity_combine(const float* __restrict__ agg_db, const float* __restrict__ deg_db,
                                  const float* __restrict__ db_root, const float* __restrict__ db_bias,
                                  const float* __restrict__ agg_rg, const float* __restrict__ deg_rg,
                                  const float* __restrict__ rg_root, const float* __restrict__ rg_bias,
                                  float* __restrict__ entity) {
    size_t i = (size_t)blockIdx.x * blockDim.x + threadIdx.x;
    if (i >= (size_t)N_ENTITY * DIM) return;
    int n = (int)(i / DIM), d = (int)(i % DIM);
    entity[i] = agg_db[i] / fmaxf(deg_db[n], 1.0f) + db_root[i] + db_bias[d]
              + agg_rg[i] / fmaxf(deg_rg[n], 1.0f) + rg_root[i] + rg_bias[d];
}

// ---------------------------------------------------------------------------
// WMMA GEMM: C[M x 128] = A[M x 128] @ Bn^T, Bn [128 x 128] stored N-major.
// 8 waves per block cover N=128; grid.x tiles M.  Optional row gather via idx.
// ---------------------------------------------------------------------------
__global__ void cm_gemm_xw(const float* __restrict__ A, const float* __restrict__ Bn,
                           float* __restrict__ C, int M) {
    int wave = threadIdx.x >> 5;
    int lane = threadIdx.x & 31;
    int n0 = wave * 16;
    int m0 = blockIdx.x * 16;
    int row = m0 + (lane & 15);
    if (row >= M) row = M - 1;                    // clamp (stores guarded)
    const float* arow = A  + (size_t)row * DIM;
    const float* brow = Bn + (size_t)(n0 + (lane & 15)) * DIM;
    v8f c = {0.f, 0.f, 0.f, 0.f, 0.f, 0.f, 0.f, 0.f};
    for (int k = 0; k < DIM; k += KSTEP)
        c = cm_mac(cm_load_frag(arow, k, lane), cm_load_frag(brow, k, lane), c);
    int col = n0 + (lane & 15);
    int rbase = (lane >> 4) * 8;
    #pragma unroll
    for (int i = 0; i < 8; ++i) {
        int m = m0 + rbase + i;
        if (m < M) C[(size_t)m * DIM + col] = c[i];
    }
}

__global__ void cm_gemm_gather_xw(const float* __restrict__ emb, const int* __restrict__ idx,
                                  const float* __restrict__ Bn,
                                  float* __restrict__ C, int M) {
    int wave = threadIdx.x >> 5;
    int lane = threadIdx.x & 31;
    int n0 = wave * 16;
    int m0 = blockIdx.x * 16;
    int row = m0 + (lane & 15);
    if (row >= M) row = M - 1;
    const float* arow = emb + (size_t)idx[row] * DIM;
    const float* brow = Bn  + (size_t)(n0 + (lane & 15)) * DIM;
    v8f c = {0.f, 0.f, 0.f, 0.f, 0.f, 0.f, 0.f, 0.f};
    for (int k = 0; k < DIM; k += KSTEP)
        c = cm_mac(cm_load_frag(arow, k, lane), cm_load_frag(brow, k, lane), c);
    int col = n0 + (lane & 15);
    int rbase = (lane >> 4) * 8;
    #pragma unroll
    for (int i = 0; i < 8; ++i) {
        int m = m0 + rbase + i;
        if (m < M) C[(size_t)m * DIM + col] = c[i];
    }
}

__global__ void cm_transpose128(const float* __restrict__ W, float* __restrict__ Wt) {
    int i = blockIdx.x * blockDim.x + threadIdx.x;   // i over 128*128
    if (i >= DIM * DIM) return;
    int k = i / DIM, n = i % DIM;
    Wt[n * DIM + k] = W[k * DIM + n];
}

// ---------------------------------------------------------------------------
// Concept GCN pieces.
// ---------------------------------------------------------------------------
__global__ void cm_con_deg_init(float* __restrict__ deg) {
    int i = blockIdx.x * blockDim.x + threadIdx.x;
    if (i < N_CONCEPT) deg[i] = 1.0f;             // self loop
}
__global__ void cm_con_deg_edge(const int* __restrict__ dst, float* __restrict__ deg, int nE) {
    int e = blockIdx.x * blockDim.x + threadIdx.x;
    if (e < nE) atomicAdd(deg + dst[e], 1.0f);
}
__global__ void cm_con_dinv(float* __restrict__ deg) {
    int i = blockIdx.x * blockDim.x + threadIdx.x;
    if (i < N_CONCEPT) deg[i] = rsqrtf(fmaxf(deg[i], 1.0f));
}
__global__ void cm_gcn_edge(const int* __restrict__ src, const int* __restrict__ dst,
                            const float* __restrict__ dinv, const float* __restrict__ h,
                            float* __restrict__ out, int nE) {
    int wave = threadIdx.x >> 5;
    int lane = threadIdx.x & 31;
    int e = blockIdx.x * 8 + wave;
    if (e >= nE) return;
    int s = src[e], dv = dst[e];
    float coef = dinv[s] * dinv[dv];
    int d0 = lane * 4;
    const float4 v = *(const float4*)(h + (size_t)s * DIM + d0);
    float* o = out + (size_t)dv * DIM + d0;
    atomicAdd(o + 0, coef * v.x); atomicAdd(o + 1, coef * v.y);
    atomicAdd(o + 2, coef * v.z); atomicAdd(o + 3, coef * v.w);
}
__global__ void cm_gcn_finalize(const float* __restrict__ dinv, const float* __restrict__ h,
                                const float* __restrict__ bias, float* __restrict__ out) {
    size_t i = (size_t)blockIdx.x * blockDim.x + threadIdx.x;
    if (i >= (size_t)N_CONCEPT * DIM) return;
    int n = (int)(i / DIM), d = (int)(i % DIM);
    out[i] += dinv[n] * dinv[n] * h[i] + bias[d];  // self-loop term + bias
}

// ---------------------------------------------------------------------------
// Attention: proj computed by gathered WMMA GEMM, then e[bl] = tanh(proj).ba,
// then masked softmax pool.
// ---------------------------------------------------------------------------
__global__ void cm_attn_reduce(const float* __restrict__ proj, const float* __restrict__ ba,
                               float* __restrict__ e_out) {
    int bl = blockIdx.x;
    int d  = threadIdx.x;                // 128 threads
    __shared__ float red[DIM];
    red[d] = tanhf(proj[(size_t)bl * DIM + d]) * ba[d];
    __syncthreads();
    for (int s = DIM / 2; s > 0; s >>= 1) {
        if (d < s) red[d] += red[d + s];
        __syncthreads();
    }
    if (d == 0) e_out[bl] = red[0];
}

__global__ void cm_attn_pool(const float* __restrict__ emb, const int* __restrict__ idx,
                             const int* __restrict__ mask, const float* __restrict__ e_in,
                             float* __restrict__ pooled, int L) {
    int b = blockIdx.x;
    int d = threadIdx.x;                 // 128 threads
    __shared__ float a[L_CON];
    __shared__ float mx, sm;
    if (d < L) a[d] = e_in[b * L + d] + NEG_INF_F * (float)mask[b * L + d];
    __syncthreads();
    if (d == 0) {
        float mm = a[0];
        for (int l = 1; l < L; ++l) mm = fmaxf(mm, a[l]);
        mx = mm;
    }
    __syncthreads();
    if (d < L) a[d] = expf(a[d] - mx);
    __syncthreads();
    if (d == 0) {
        float s = 0.f;
        for (int l = 0; l < L; ++l) s += a[l];
        sm = s;
    }
    __syncthreads();
    float acc = 0.f;
    for (int l = 0; l < L; ++l) {
        int node = idx[b * L + l];
        acc += a[l] * emb[(size_t)node * DIM + d];
    }
    pooled[b * DIM + d] = acc / sm;
}

// ---------------------------------------------------------------------------
// Gate fusion: uc = [udb|ucon] @ Wn + bn ; g = sigmoid(uc . gW + gb)
// ---------------------------------------------------------------------------
__global__ void cm_gate(const float* __restrict__ user_db, const float* __restrict__ user_con,
                        const float* __restrict__ Wn, const float* __restrict__ bn,
                        const float* __restrict__ gW, const float* __restrict__ gb,
                        float* __restrict__ user) {
    int b = blockIdx.x;
    int d = threadIdx.x;                 // 128 threads
    __shared__ float udb[DIM], ucon[DIM], red[DIM];
    __shared__ float gsh;
    udb[d]  = user_db[b * DIM + d];
    ucon[d] = user_con[b * DIM + d];
    __syncthreads();
    float uc = bn[d];
    #pragma unroll 4
    for (int k = 0; k < DIM; ++k) uc += udb[k] * Wn[k * DIM + d];
    #pragma unroll 4
    for (int k = 0; k < DIM; ++k) uc += ucon[k] * Wn[(DIM + k) * DIM + d];
    red[d] = uc * gW[d];
    __syncthreads();
    for (int s = DIM / 2; s > 0; s >>= 1) {
        if (d < s) red[d] += red[d + s];
        __syncthreads();
    }
    if (d == 0) gsh = 1.f / (1.f + expf(-(red[0] + gb[0])));
    __syncthreads();
    float g = gsh;
    user[b * DIM + d] = g * udb[d] + (1.f - g) * ucon[d];
}

// ---------------------------------------------------------------------------
// scores[128 x N_ENTITY] = user @ entity^T + out_bias.
// One wave owns a 16-wide N strip and ALL 8 M-tiles (8 accumulators): each B
// fragment (entity) is loaded once and reused by 8 WMMAs, so the 33MB entity
// table is streamed exactly once.  N_ENTITY = 16*4023 -> live strips are full.
// ---------------------------------------------------------------------------
__global__ void cm_scores(const float* __restrict__ U, const float* __restrict__ E,
                          const float* __restrict__ bias, float* __restrict__ out) {
    int wave = threadIdx.x >> 5;
    int lane = threadIdx.x & 31;
    int n0 = blockIdx.x * 128 + wave * 16;
    if (n0 >= N_ENTITY) return;          // wave-uniform: EXEC stays all-ones
    int mr = lane & 15;
    const float* brow = E + (size_t)(n0 + mr) * DIM;
    v8f c[8];
    #pragma unroll
    for (int t = 0; t < 8; ++t)
        c[t] = (v8f){0.f, 0.f, 0.f, 0.f, 0.f, 0.f, 0.f, 0.f};
    for (int k = 0; k < DIM; k += KSTEP) {
        frag_t b = cm_load_frag(brow, k, lane);
        #pragma unroll
        for (int t = 0; t < 8; ++t) {
            const float* arow = U + (size_t)(t * 16 + mr) * DIM;
            c[t] = cm_mac(cm_load_frag(arow, k, lane), b, c[t]);
        }
    }
    int col = n0 + mr;
    float bz = bias[col];
    int rbase = (lane >> 4) * 8;
    #pragma unroll
    for (int t = 0; t < 8; ++t)
        #pragma unroll
        for (int i = 0; i < 8; ++i)
            out[(size_t)(t * 16 + rbase + i) * N_ENTITY + col] = c[t][i] + bz;
}

// ---------------------------------------------------------------------------
extern "C" void kernel_launch(void* const* d_in, const int* in_sizes, int n_in,
                              void* d_out, int out_size, void* d_ws, size_t ws_size,
                              hipStream_t stream) {
    const int*   db_ei     = (const int*)  d_in[0];
    const int*   db_et     = (const int*)  d_in[1];
    const int*   rg_ei     = (const int*)  d_in[2];
    const int*   rg_et     = (const int*)  d_in[3];
    const int*   con_ei    = (const int*)  d_in[4];
    const int*   db_idx    = (const int*)  d_in[5];
    const int*   db_mask   = (const int*)  d_in[6];
    const int*   con_idx   = (const int*)  d_in[7];
    const int*   con_mask  = (const int*)  d_in[8];
    const float* db_basis  = (const float*)d_in[9];
    const float* db_comp   = (const float*)d_in[10];
    const float* db_root   = (const float*)d_in[11];
    const float* db_bias   = (const float*)d_in[12];
    const float* rg_basis  = (const float*)d_in[13];
    const float* rg_comp   = (const float*)d_in[14];
    const float* rg_root   = (const float*)d_in[15];
    const float* rg_bias   = (const float*)d_in[16];
    const float* conc_emb  = (const float*)d_in[17];
    const float* gcn_W     = (const float*)d_in[18];
    const float* gcn_b     = (const float*)d_in[19];
    const float* Wa_db     = (const float*)d_in[20];
    const float* ba_db     = (const float*)d_in[21];
    const float* Wa_con    = (const float*)d_in[22];
    const float* ba_con    = (const float*)d_in[23];
    const float* un_W      = (const float*)d_in[24];
    const float* un_b      = (const float*)d_in[25];
    const float* gn_W      = (const float*)d_in[26];
    const float* gn_b      = (const float*)d_in[27];
    const float* out_bias  = (const float*)d_in[28];

    // ---- workspace layout (floats) -- zeroed region first -----------------
    float* ws = (float*)d_ws;
    size_t off = 0;
    const size_t ND  = (size_t)N_ENTITY * DIM;
    const size_t NCD = (size_t)N_CONCEPT * DIM;
    float* agg_db   = ws + off; off += ND;
    float* deg_db   = ws + off; off += N_ENTITY;
    float* agg_rg   = ws + off; off += ND;
    float* deg_rg   = ws + off; off += N_ENTITY;
    float* con_out  = ws + off; off += NCD;
    size_t zero_floats = off;
    float* entity   = ws + off; off += ND;
    float* h0       = ws + off; off += NCD;
    float* con_dinv = ws + off; off += N_CONCEPT;
    float* Wt_gcn   = ws + off; off += DIM * DIM;
    float* Wt_db    = ws + off; off += DIM * DIM;
    float* Wt_con   = ws + off; off += DIM * DIM;
    float* proj_db  = ws + off; off += (size_t)B_SZ * L_DB * DIM;
    float* proj_con = ws + off; off += (size_t)B_SZ * L_CON * DIM;
    float* e_db     = ws + off; off += B_SZ * L_DB;
    float* e_con    = ws + off; off += B_SZ * L_CON;
    float* user_db  = ws + off; off += B_SZ * DIM;
    float* user_con = ws + off; off += B_SZ * DIM;
    float* user     = ws + off; off += B_SZ * DIM;
    size_t base_floats = off;
    float* weightbuf = ws + off;                     // optional [RD_DB, N, D]
    const size_t weight_floats = (size_t)RD_DB * ND;
    const bool use_w = ws_size >= (base_floats + weight_floats) * sizeof(float);

    hipMemsetAsync(ws, 0, zero_floats * sizeof(float), stream);

    // ---- entity RGCN (db + rg) -------------------------------------------
    if (use_w) {
        cm_make_weight<<<(int)((ND / 4 + 255) / 256), 256, 0, stream>>>(
            db_basis, db_comp, weightbuf, RD_DB);
        cm_rgcn_edge_w<<<(NE_DB + 7) / 8, 256, 0, stream>>>(
            db_ei, db_ei + NE_DB, db_et, weightbuf, agg_db, deg_db, NE_DB);
        cm_make_weight<<<(int)((ND / 4 + 255) / 256), 256, 0, stream>>>(
            rg_basis, rg_comp, weightbuf, RD_RG);
        cm_rgcn_edge_w<<<(NE_RG + 7) / 8, 256, 0, stream>>>(
            rg_ei, rg_ei + NE_RG, rg_et, weightbuf, agg_rg, deg_rg, NE_RG);
    } else {
        cm_rgcn_edge<<<(NE_DB + 7) / 8, 256, 0, stream>>>(
            db_ei, db_ei + NE_DB, db_et, db_basis, db_comp, agg_db, deg_db, NE_DB);
        cm_rgcn_edge<<<(NE_RG + 7) / 8, 256, 0, stream>>>(
            rg_ei, rg_ei + NE_RG, rg_et, rg_basis, rg_comp, agg_rg, deg_rg, NE_RG);
    }
    cm_entity_combine<<<(int)((ND + 255) / 256), 256, 0, stream>>>(
        agg_db, deg_db, db_root, db_bias, agg_rg, deg_rg, rg_root, rg_bias, entity);

    // ---- concept GCN ------------------------------------------------------
    cm_transpose128<<<(DIM * DIM + 255) / 256, 256, 0, stream>>>(gcn_W, Wt_gcn);
    cm_gemm_xw<<<(N_CONCEPT + 15) / 16, 256, 0, stream>>>(conc_emb, Wt_gcn, h0, N_CONCEPT);
    cm_con_deg_init<<<(N_CONCEPT + 255) / 256, 256, 0, stream>>>(con_dinv);
    cm_con_deg_edge<<<(NE_CON + 255) / 256, 256, 0, stream>>>(con_ei + NE_CON, con_dinv, NE_CON);
    cm_con_dinv<<<(N_CONCEPT + 255) / 256, 256, 0, stream>>>(con_dinv);
    cm_gcn_edge<<<(NE_CON + 7) / 8, 256, 0, stream>>>(
        con_ei, con_ei + NE_CON, con_dinv, h0, con_out, NE_CON);
    cm_gcn_finalize<<<(int)((NCD + 255) / 256), 256, 0, stream>>>(con_dinv, h0, gcn_b, con_out);

    // ---- attention pooling (WMMA projection + reduce + softmax pool) ------
    cm_transpose128<<<(DIM * DIM + 255) / 256, 256, 0, stream>>>(Wa_db, Wt_db);
    cm_gemm_gather_xw<<<(B_SZ * L_DB + 15) / 16, 256, 0, stream>>>(
        entity, db_idx, Wt_db, proj_db, B_SZ * L_DB);
    cm_attn_reduce<<<B_SZ * L_DB, DIM, 0, stream>>>(proj_db, ba_db, e_db);
    cm_attn_pool<<<B_SZ, DIM, 0, stream>>>(entity, db_idx, db_mask, e_db, user_db, L_DB);

    cm_transpose128<<<(DIM * DIM + 255) / 256, 256, 0, stream>>>(Wa_con, Wt_con);
    cm_gemm_gather_xw<<<(B_SZ * L_CON + 15) / 16, 256, 0, stream>>>(
        con_out, con_idx, Wt_con, proj_con, B_SZ * L_CON);
    cm_attn_reduce<<<B_SZ * L_CON, DIM, 0, stream>>>(proj_con, ba_con, e_con);
    cm_attn_pool<<<B_SZ, DIM, 0, stream>>>(con_out, con_idx, con_mask, e_con, user_con, L_CON);

    // ---- gate fusion ------------------------------------------------------
    cm_gate<<<B_SZ, DIM, 0, stream>>>(user_db, user_con, un_W, un_b, gn_W, gn_b, user);

    // ---- final scores GEMM (WMMA, B reused across 8 M-tiles) --------------
    cm_scores<<<(N_ENTITY + 127) / 128, 256, 0, stream>>>(
        user, entity, out_bias, (float*)d_out);
}
